// ConvLSTMModel_89283780149902
// MI455X (gfx1250) — compile-verified, matrix operand order checked by reading, not run on hardware
//
#include <hip/hip_runtime.h>

// ---------------------------------------------------------------------------
// ConvLSTM (B=16, TS=16, N=4096, FC=1, HIDDEN=64) for gfx1250 / MI455X.
// Gate conv done as bf16 WMMA (v_wmma_f32_16x16x32_bf16), fp32 state update.
// ---------------------------------------------------------------------------

typedef __attribute__((ext_vector_type(16))) __bf16 v16bf;
typedef __attribute__((ext_vector_type(8)))  float  v8f;

#define NNE  4096   // spatial positions
#define BBA  16     // batch
#define TSN  16     // time steps
#define HID  64     // hidden channels
#define CIN  65     // 1 (x) + HID
#define GOUT 256    // 4*HID gate outputs
#define AFRAG_ELEMS (16*3*2*512)   // 16 gate tiles * 3 taps * 2 ktiles * 32 lanes * 16 bf16

union Frag {
    uint4  q[2];
    v16bf  v;
};

__device__ __forceinline__ unsigned short f2bf(float f) {
    unsigned int u = __builtin_bit_cast(unsigned int, f);
    u += 0x7fffu + ((u >> 16) & 1u);           // round-to-nearest-even
    return (unsigned short)(u >> 16);
}
__device__ __forceinline__ float bf2f(unsigned short s) {
    unsigned int u = ((unsigned int)s) << 16;
    return __builtin_bit_cast(float, u);
}
__device__ __forceinline__ float sigm(float x) {
    return 1.0f / (1.0f + __expf(-x));
}
__device__ __forceinline__ float tanh_fast(float x) {
    return 1.0f - 2.0f / (__expf(2.0f * x) + 1.0f);
}

// ---------------------------------------------------------------------------
// Prep: permute conv weights into WMMA A-fragment order (bf16), and build the
// per-gate constants {bias, wx_tap0, wx_tap1, wx_tap2} (x is input channel 0;
// only the center kernel column kw=1 matters since width==1 with padding 1).
// ---------------------------------------------------------------------------
__global__ __launch_bounds__(256) void prep_kernel(
    const float* __restrict__ conv_w, const float* __restrict__ conv_b,
    unsigned short* __restrict__ afrag, float4* __restrict__ gparams) {
    int id = blockIdx.x * 256 + threadIdx.x;
    if (id < AFRAG_ELEMS) {
        int j    = id & 15;
        int lane = (id >> 4) & 31;
        int kt   = (id >> 9) & 1;
        int tap  = (id >> 10) % 3;
        int gt   = id / 3072;
        // 16-bit A layout: lanes 0-15 -> M=lane, K=0..7,16..23; lanes 16-31 -> +8
        int M  = lane & 15;
        int K  = (j < 8 ? j : j + 8) + ((lane >> 4) ? 8 : 0);
        int ch = kt * 32 + K;                       // h channel 0..63
        int o  = gt * 16 + M;                       // gate output 0..255
        float val = conv_w[(size_t)o * (CIN * 9) + (size_t)(1 + ch) * 9 + tap * 3 + 1];
        afrag[id] = f2bf(val);
    } else if (id < AFRAG_ELEMS + GOUT) {
        int o = id - AFRAG_ELEMS;
        gparams[o] = make_float4(conv_b[o],
                                 conv_w[(size_t)o * (CIN * 9) + 1],   // ch0, tap0
                                 conv_w[(size_t)o * (CIN * 9) + 4],   // ch0, tap1
                                 conv_w[(size_t)o * (CIN * 9) + 7]);  // ch0, tap2
    }
}

__global__ __launch_bounds__(256) void zero_kernel(uint4* __restrict__ p, int n4) {
    int id = blockIdx.x * 256 + threadIdx.x;
    if (id < n4) p[id] = make_uint4(0u, 0u, 0u, 0u);
}

// ---------------------------------------------------------------------------
// One recurrence step: gates = W_h (*) h_prev (3-tap, bf16 WMMA) + wx (*) x + b
//   c = sig(f)*c + sig(i)*tanh(g);  h = sig(o)*tanh(c)
// Grid: (N/64, B); 256 threads = 8 wave32 waves.
// Wave w: position tile pt = w&3 (16 cols), hidden tiles ht = (w>>2)*2 + {0,1}.
// ---------------------------------------------------------------------------
__global__ __launch_bounds__(256) void convlstm_step(
    const float* __restrict__ xG,
    const unsigned short* __restrict__ hprev,   // [b][n][ch] bf16
    unsigned short* __restrict__ hnext,
    float* __restrict__ cbuf,                   // [b][n][ch] fp32
    const unsigned short* __restrict__ afrag,
    const float4* __restrict__ gparams,
    int t) {
    __shared__ __align__(16) unsigned short hS[66 * 64];  // halo tile, bf16
    __shared__ float xS[66];

    const int tid = threadIdx.x;
    const int b   = blockIdx.y;
    const int n0  = blockIdx.x * 64;

    // ---- stage h tile (66 rows x 64 ch) into LDS, zero-padded at edges
    for (int q = tid; q < 66 * 8; q += 256) {     // 16B chunks (8 bf16)
        int r = q >> 3, c8 = q & 7;
        int n = n0 - 1 + r;
        uint4 val = make_uint4(0u, 0u, 0u, 0u);
        if (n >= 0 && n < NNE)
            val = ((const uint4*)hprev)[(size_t)(b * NNE + n) * 8 + c8];
        ((uint4*)hS)[q] = val;
    }
    if (tid < 66) {
        int n = n0 - 1 + tid;
        xS[tid] = (n >= 0 && n < NNE) ? xG[((size_t)(b * TSN + t)) * NNE + n] : 0.0f;
    }
    __syncthreads();

    const int lane   = tid & 31;
    const int wid    = tid >> 5;
    const int pt     = wid & 3;
    const int htbase = (wid >> 2) * 2;
    const int col    = lane & 15;
    const int half8  = (lane >> 4) ? 8 : 0;

    // ---- build 6 B fragments (3 taps x 2 K-tiles) for this wave's 16 columns
    Frag bfr[6];
    #pragma unroll
    for (int tap = 0; tap < 3; ++tap) {
        #pragma unroll
        for (int kt = 0; kt < 2; ++kt) {
            const int pos  = pt * 16 + col + tap;   // LDS row (includes halo -1)
            const int base = kt * 32 + half8;       // K chunk start
            Frag f;
            f.q[0] = *(const uint4*)&hS[pos * 64 + base];        // K=base..base+7
            f.q[1] = *(const uint4*)&hS[pos * 64 + base + 16];   // K=base+16..+23
            bfr[tap * 2 + kt] = f;
        }
    }

    const uint4* afq = (const uint4*)afrag;
    const float xm1 = xS[pt * 16 + col];        // x[n-1]
    const float x0  = xS[pt * 16 + col + 1];    // x[n]
    const float xp1 = xS[pt * 16 + col + 2];    // x[n+1]
    const int   n   = n0 + pt * 16 + col;

    #pragma unroll
    for (int hh = 0; hh < 2; ++hh) {
        const int ht = htbase + hh;
        v8f acc[4];
        #pragma unroll
        for (int gg = 0; gg < 4; ++gg) {        // i, f, o, g gate groups
            v8f c = {};
            const int gt = gg * 4 + ht;
            #pragma unroll
            for (int f6 = 0; f6 < 6; ++f6) {    // tap*2 + ktile
                Frag a;
                const int fi = gt * 6 + f6;
                a.q[0] = afq[fi * 64 + lane * 2 + 0];
                a.q[1] = afq[fi * 64 + lane * 2 + 1];
                c = __builtin_amdgcn_wmma_f32_16x16x32_bf16(
                        false, a.v, false, bfr[f6].v, (short)0, c, false, false);
            }
            acc[gg] = c;
        }

        // ---- elementwise LSTM update; rows map to 8 consecutive hidden ch
        const size_t cbase = ((size_t)(b * NNE + n)) * HID + ht * 16 + half8;
        const float4* cp = (const float4*)(cbuf + cbase);
        float4 c0v = cp[0], c1v = cp[1];
        float cOld[8] = {c0v.x, c0v.y, c0v.z, c0v.w, c1v.x, c1v.y, c1v.z, c1v.w};
        float cNew[8], hNew[8];
        #pragma unroll
        for (int v = 0; v < 8; ++v) {
            const int row = half8 + v;
            float g4[4];
            #pragma unroll
            for (int gg = 0; gg < 4; ++gg) {
                const int o = (gg * 4 + ht) * 16 + row;
                float4 gp = gparams[o];
                g4[gg] = acc[gg][v] + gp.x + gp.y * xm1 + gp.z * x0 + gp.w * xp1;
            }
            float I = sigm(g4[0]), F = sigm(g4[1]);
            float O = sigm(g4[2]), G = tanh_fast(g4[3]);
            float c2 = F * cOld[v] + I * G;
            cNew[v] = c2;
            hNew[v] = O * tanh_fast(c2);
        }
        float4* cwp = (float4*)(cbuf + cbase);
        cwp[0] = make_float4(cNew[0], cNew[1], cNew[2], cNew[3]);
        cwp[1] = make_float4(cNew[4], cNew[5], cNew[6], cNew[7]);
        uint4 hp;
        hp.x = (unsigned)f2bf(hNew[0]) | ((unsigned)f2bf(hNew[1]) << 16);
        hp.y = (unsigned)f2bf(hNew[2]) | ((unsigned)f2bf(hNew[3]) << 16);
        hp.z = (unsigned)f2bf(hNew[4]) | ((unsigned)f2bf(hNew[5]) << 16);
        hp.w = (unsigned)f2bf(hNew[6]) | ((unsigned)f2bf(hNew[7]) << 16);
        ((uint4*)hnext)[(size_t)(b * NNE + n) * 8 + ht * 2 + (half8 >> 3)] = hp;
    }
}

// ---------------------------------------------------------------------------
// Final 1x1 conv: out[b,ts,n] = dot(h_final[b,n,:], fc_w) + fc_b, bcast over ts
// ---------------------------------------------------------------------------
__global__ __launch_bounds__(256) void fc_out_kernel(
    const unsigned short* __restrict__ hfin,
    const float* __restrict__ fcw, const float* __restrict__ fcb,
    float* __restrict__ out) {
    int id = blockIdx.x * 256 + threadIdx.x;   // (b,n) flat
    if (id >= BBA * NNE) return;
    int b = id >> 12;
    int n = id & (NNE - 1);
    const uint4* hp = (const uint4*)hfin + (size_t)id * 8;
    float acc = 0.0f;
    #pragma unroll
    for (int q = 0; q < 8; ++q) {
        uint4 u = hp[q];
        unsigned int w[4] = {u.x, u.y, u.z, u.w};
        #pragma unroll
        for (int k = 0; k < 4; ++k) {
            int c = q * 8 + k * 2;
            acc += bf2f((unsigned short)(w[k] & 0xffffu)) * fcw[c];
            acc += bf2f((unsigned short)(w[k] >> 16))     * fcw[c + 1];
        }
    }
    float r = acc + fcb[0];
    #pragma unroll
    for (int ts = 0; ts < TSN; ++ts)
        out[((size_t)(b * TSN + ts)) * NNE + n] = r;
}

// ---------------------------------------------------------------------------
extern "C" void kernel_launch(void* const* d_in, const int* in_sizes, int n_in,
                              void* d_out, int out_size, void* d_ws, size_t ws_size,
                              hipStream_t stream) {
    (void)in_sizes; (void)n_in; (void)out_size; (void)ws_size;
    const float* x      = (const float*)d_in[0];
    const float* conv_w = (const float*)d_in[1];
    const float* conv_b = (const float*)d_in[2];
    const float* fc_w   = (const float*)d_in[3];
    const float* fc_b   = (const float*)d_in[4];
    float* out = (float*)d_out;

    char* ws = (char*)d_ws;
    const size_t szA  = (size_t)AFRAG_ELEMS * 2;            // 96 KB bf16 A-frags
    const size_t szG  = (size_t)GOUT * sizeof(float4);      // per-gate consts
    const size_t szH  = (size_t)BBA * NNE * HID * 2;        // 8 MB bf16 h
    const size_t szC  = (size_t)BBA * NNE * HID * 4;        // 16 MB fp32 c
    unsigned short* afrag = (unsigned short*)(ws);
    float4*         gpar  = (float4*)(ws + szA);
    unsigned short* h0    = (unsigned short*)(ws + szA + szG + 256);  // pad/align
    unsigned short* h1    = (unsigned short*)((char*)h0 + szH);
    float*          cbuf  = (float*)((char*)h1 + szH);

    prep_kernel<<<(AFRAG_ELEMS + GOUT + 255) / 256, 256, 0, stream>>>(
        conv_w, conv_b, afrag, gpar);

    const int h4 = (int)(szH / 16), c4 = (int)(szC / 16);
    zero_kernel<<<(h4 + 255) / 256, 256, 0, stream>>>((uint4*)h0, h4);
    zero_kernel<<<(c4 + 255) / 256, 256, 0, stream>>>((uint4*)cbuf, c4);

    unsigned short* hb[2] = {h0, h1};
    for (int t = 0; t < TSN; ++t) {
        convlstm_step<<<dim3(NNE / 64, BBA), 256, 0, stream>>>(
            x, hb[t & 1], hb[(t + 1) & 1], cbuf, afrag, gpar, t);
    }
    // t=15 wrote hb[0] == h0 -> final hidden state
    fc_out_kernel<<<(BBA * NNE + 255) / 256, 256, 0, stream>>>(h0, fc_w, fc_b, out);
}
